// Attention_85693187490059
// MI455X (gfx1250) — matrix operand order repaired and checked by reading, not compiled, not run on hardware
//
#include <hip/hip_runtime.h>
#include <hip/hip_bf16.h>

// ---------------------------------------------------------------------------
// Axial attention block for MI455X (gfx1250, wave32, WMMA).
// All three matmul stages run on v_wmma_f32_16x16x32_bf16 (f32 accumulate).
// Weight tiles are staged into LDS with gfx1250 async global->LDS copies
// (double buffered, ASYNCcnt-tracked).
// ---------------------------------------------------------------------------

static constexpr int kHeads   = 16;
static constexpr int kDimHead = 64;
static constexpr int kBatch   = 4;
static constexpr int kFrames  = 16;
static constexpr int kNseq    = 196;                  // tokens per frame
static constexpr int kNtok    = 1 + kFrames * kNseq;  // 3137
static constexpr int kDim     = 1024;
static constexpr int kM       = kBatch * kNtok;       // 12548 rows
static constexpr int kNkey    = 1 + kNseq;            // 197 (cls + frame)
static constexpr int kNkeyP   = 224;                  // padded to 7*32
static constexpr int kBStride = 40;                   // LDS B-tile row stride

typedef __bf16 bf16_t;
typedef __attribute__((ext_vector_type(2)))  __bf16 bf16x2;
typedef __attribute__((ext_vector_type(8)))  __bf16 v8bf;
typedef __attribute__((ext_vector_type(16))) __bf16 v16bf;
typedef __attribute__((ext_vector_type(8)))  float  v8f;
typedef __attribute__((ext_vector_type(4)))  int    v4i;

#if __has_builtin(__builtin_amdgcn_global_load_async_to_lds_b128) && \
    __has_builtin(__builtin_amdgcn_s_wait_asynccnt)
#define USE_ASYNC_LDS 1
typedef __attribute__((address_space(1))) void as1_void;
typedef __attribute__((address_space(3))) void as3_void;
typedef __attribute__((address_space(1))) v4i  as1_v4i;
typedef __attribute__((address_space(3))) v4i  as3_v4i;
// generic -> AS void* (proven to compile) -> same-AS pointee cast
#define AS1_V4I(p) ((as1_v4i*)(as1_void*)(p))
#define AS3_V4I(p) ((as3_v4i*)(as3_void*)(p))
#else
#define USE_ASYNC_LDS 0
#endif

__device__ __forceinline__ v8f wmma_bf16(v16bf a, v16bf b, v8f c) {
  // (neg_a, A, neg_b, B, c_mod, C, reuse_a, reuse_b)
  return __builtin_amdgcn_wmma_f32_16x16x32_bf16(false, a, false, b,
                                                 (short)0, c, false, false);
}

// Stage a 64-col x 32-K bf16 B tile from wT (N x K, row major) into LDS.
// 4096 B total -> 256 x 16 B chunks -> 2 chunks per thread (128 threads).
__device__ __forceinline__ void stage_b_tile(const bf16_t* __restrict__ wT,
                                             int colBase, int k0,
                                             bf16_t* sB, int tid) {
#pragma unroll
  for (int c = tid; c < 256; c += 128) {
    const int col  = c >> 2;
    const int part = c & 3;
    const bf16_t* src = wT + (size_t)(colBase + col) * kDim + k0 + part * 8;
    bf16_t* dst = sB + col * kBStride + part * 8;
#if USE_ASYNC_LDS
    __builtin_amdgcn_global_load_async_to_lds_b128(AS1_V4I(src), AS3_V4I(dst),
                                                   0, 0);
#else
    *(uint4*)dst = *(const uint4*)src;
#endif
  }
}

// Read one 32x16 B fragment (k = 8*hlf..+7 and 16+8*hlf..+7 per lane) from the LDS tile.
__device__ __forceinline__ v16bf load_b_frag_lds(const bf16_t* sB, int col,
                                                 int hlf) {
  const v8bf* p = (const v8bf*)(sB + (size_t)col * kBStride + hlf * 8);
  v8bf lo = p[0], hi = p[2];
  v16bf b;
#pragma unroll
  for (int e = 0; e < 8; ++e) { b[e] = lo[e]; b[8 + e] = hi[e]; }
  return b;
}

// ---------------------------------------------------------------------------
// Transpose fp32 KxN weight into bf16 NxK (so WMMA B-fragments are
// contiguous loads).
// ---------------------------------------------------------------------------
__global__ __launch_bounds__(256) void transpose_w_kernel(
    const float* __restrict__ in, bf16_t* __restrict__ outT, int K, int N) {
  __shared__ float tile[32][33];
  const int kb = blockIdx.x * 32;
  const int nb = blockIdx.y * 32;
  const int tx = threadIdx.x & 31;
  const int ty = threadIdx.x >> 5;  // 0..7
  for (int i = ty; i < 32; i += 8)
    tile[i][tx] = in[(size_t)(kb + i) * N + (nb + tx)];
  __syncthreads();
  for (int i = ty; i < 32; i += 8)
    outT[(size_t)(nb + i) * K + (kb + tx)] = (bf16_t)tile[tx][i];
}

// ---------------------------------------------------------------------------
// QKV GEMM: (12548 x 1024) fp32  @  wqkvT (3072 x 1024) bf16 -> q/k/v bf16
// Block: 128 threads (4 waves); block tile 64x64; wave tile 16x64.
// B tile shared by all 4 waves via double-buffered async LDS staging.
// Epilogue fuses head split + q scale (dh^-0.5) + layout [B*H][ntok][64].
// ---------------------------------------------------------------------------
__global__ __launch_bounds__(128) void gemm_qkv_kernel(
    const float* __restrict__ x, const bf16_t* __restrict__ wT,
    bf16_t* __restrict__ qb, bf16_t* __restrict__ kb, bf16_t* __restrict__ vb) {
  __shared__ bf16_t sB[2][64 * kBStride];

  const int tid   = threadIdx.x;
  const int lane  = tid & 31;
  const int wave  = tid >> 5;
  const int mlane = lane & 15;
  const int hlf   = lane >> 4;
  const int row0  = blockIdx.x * 64 + wave * 16;
  const int col0  = blockIdx.y * 64;

  int arow = row0 + mlane;
  if (arow >= kM) arow = kM - 1;
  const float* aptr = x + (size_t)arow * kDim;

  const v8f zero8 = {0.f, 0.f, 0.f, 0.f, 0.f, 0.f, 0.f, 0.f};
  v8f acc[4];
#pragma unroll
  for (int j = 0; j < 4; ++j) acc[j] = zero8;

  stage_b_tile(wT, col0, 0, sB[0], tid);

  for (int kt = 0; kt < kDim / 32; ++kt) {
    const int k0  = kt * 32;
    const int buf = kt & 1;
    if (kt + 1 < kDim / 32) {
      stage_b_tile(wT, col0, k0 + 32, sB[buf ^ 1], tid);
#if USE_ASYNC_LDS
      __builtin_amdgcn_s_wait_asynccnt(2);  // tile for this iter complete
#endif
    } else {
#if USE_ASYNC_LDS
      __builtin_amdgcn_s_wait_asynccnt(0);
#endif
    }
    __syncthreads();

    if (k0 + 256 < kDim) __builtin_prefetch(aptr + k0 + 256, 0, 0);
    // A fragment: fp32 -> bf16 on the fly.  Lane = M, ISA 16-bit A layout.
    v16bf afrag;
#pragma unroll
    for (int v = 0; v < 8; ++v) {
      const int kk = ((v & 4) ? 16 : 0) + hlf * 8 + (v & 3) * 2;
      float2 p = *(const float2*)(aptr + k0 + kk);
      afrag[2 * v]     = (bf16_t)p.x;
      afrag[2 * v + 1] = (bf16_t)p.y;
    }
#pragma unroll
    for (int j = 0; j < 4; ++j) {
      v16bf bfrag = load_b_frag_lds(sB[buf], j * 16 + mlane, hlf);
      acc[j] = wmma_bf16(afrag, bfrag, acc[j]);
    }
    __syncthreads();
  }

  // Epilogue: split into q/k/v with head-major layout, scale q by 1/8.
#pragma unroll
  for (int j = 0; j < 4; ++j) {
    const int c     = col0 + j * 16 + mlane;
    const int which = c >> 10;  // 0=q 1=k 2=v
    const int rem   = c & 1023;
    const int head  = rem >> 6;
    const int d     = rem & 63;
    bf16_t* dst    = (which == 0) ? qb : ((which == 1) ? kb : vb);
    const float sc = (which == 0) ? 0.125f : 1.0f;
#pragma unroll
    for (int r = 0; r < 8; ++r) {
      const int row = row0 + r + 8 * hlf;
      if (row < kM) {
        const int bidx = row / kNtok;
        const int t    = row - bidx * kNtok;
        dst[((size_t)(bidx * kHeads + head) * kNtok + t) * kDimHead + d] =
            (bf16_t)(acc[j][r] * sc);
      }
    }
  }
}

// ---------------------------------------------------------------------------
// CLS attention: one query row per (b,head) over all 3137 keys. Tiny.
// ---------------------------------------------------------------------------
__global__ __launch_bounds__(64) void cls_attn_kernel(
    const bf16_t* __restrict__ qb, const bf16_t* __restrict__ kb,
    const bf16_t* __restrict__ vb, bf16_t* __restrict__ attn) {
  __shared__ float sq[kDimHead];
  __shared__ float slog[kNtok];
  __shared__ float sred[64];
  const int bh = blockIdx.x;
  const int b = bh >> 4, h = bh & 15;
  const bf16_t* qh = qb + (size_t)bh * kNtok * kDimHead;
  const bf16_t* kh = kb + (size_t)bh * kNtok * kDimHead;
  const bf16_t* vh = vb + (size_t)bh * kNtok * kDimHead;
  const int tid = threadIdx.x;

  sq[tid] = (float)qh[tid];  // cls query (already scaled)
  __syncthreads();

  float lmax = -3.0e38f;
  for (int j = tid; j < kNtok; j += 64) {
    const bf16_t* kr = kh + (size_t)j * kDimHead;
    float s = 0.f;
#pragma unroll
    for (int d = 0; d < kDimHead; ++d) s += sq[d] * (float)kr[d];
    slog[j] = s;
    lmax = fmaxf(lmax, s);
  }
  sred[tid] = lmax;
  __syncthreads();
  for (int s = 32; s > 0; s >>= 1) {
    if (tid < s) sred[tid] = fmaxf(sred[tid], sred[tid + s]);
    __syncthreads();
  }
  const float mx = sred[0];
  __syncthreads();

  float lsum = 0.f;
  for (int j = tid; j < kNtok; j += 64) {
    float p = __expf(slog[j] - mx);
    slog[j] = p;
    lsum += p;
  }
  sred[tid] = lsum;
  __syncthreads();
  for (int s = 32; s > 0; s >>= 1) {
    if (tid < s) sred[tid] += sred[tid + s];
    __syncthreads();
  }
  const float rinv = 1.0f / sred[0];

  float o = 0.f;
  for (int j = 0; j < kNtok; ++j)
    o += slog[j] * (float)vh[(size_t)j * kDimHead + tid];
  attn[((size_t)b * kNtok) * kDim + h * kDimHead + tid] = (bf16_t)(o * rinv);
}

// ---------------------------------------------------------------------------
// Axial attention: one block per (b*head, frame). 128 threads = 4 waves,
// each wave owns full 16-row query tiles (13 tiles of 196 rows).
//   S = q @ k^T  : WMMA, K^T B-fragments gathered straight from k buffer
//   softmax      : per-row, shfl_xor reductions within 16-lane half-groups
//   O = P @ V    : WMMA, P staged bf16 in per-wave LDS, V async-staged in LDS
// ---------------------------------------------------------------------------
__global__ __launch_bounds__(128) void axial_attn_kernel(
    const bf16_t* __restrict__ qb, const bf16_t* __restrict__ kb,
    const bf16_t* __restrict__ vb, bf16_t* __restrict__ attn) {
  __shared__ bf16_t sV[kNkeyP * kDimHead];  // 224*64*2 = 28 KB
  __shared__ bf16_t sP[4][16 * kNkeyP];     // 4*16*224*2 = 28 KB

  const int bh   = blockIdx.x >> 4;
  const int fidx = blockIdx.x & 15;
  const int b = bh >> 4, h = bh & 15;
  const int tid   = threadIdx.x;
  const int lane  = tid & 31;
  const int wave  = tid >> 5;
  const int mlane = lane & 15;
  const int hlf   = lane >> 4;
  const size_t base = (size_t)bh * kNtok * kDimHead;
  const bf16_t* qh = qb + base;
  const bf16_t* kh = kb + base;
  const bf16_t* vh = vb + base;

  // Stage V rows into LDS: row 0 = cls, rows 1..196 = frame, 197..223 zero.
#if USE_ASYNC_LDS
  for (int c = tid; c < kNkey * 4; c += 128) {  // 16B chunks
    const int row  = c >> 2;
    const int part = c & 3;
    const int t = (row == 0) ? 0 : (1 + fidx * kNseq + row - 1);
    __builtin_amdgcn_global_load_async_to_lds_b128(
        AS1_V4I(vh + (size_t)t * kDimHead + part * 8),
        AS3_V4I(sV + row * kDimHead + part * 8), 0, 0);
  }
  for (int c = tid; c < (kNkeyP - kNkey) * 32; c += 128) {  // zero pad rows
    const int row = kNkey + (c >> 5);
    const int cw  = c & 31;
    *(unsigned int*)(sV + row * kDimHead + cw * 2) = 0u;
  }
  __builtin_amdgcn_s_wait_asynccnt(0);
#else
  for (int idx = tid; idx < kNkeyP * (kDimHead / 2); idx += 128) {
    const int row = idx >> 5;
    const int cw  = idx & 31;
    unsigned int val = 0u;
    if (row < kNkey) {
      const int t = (row == 0) ? 0 : (1 + fidx * kNseq + row - 1);
      val = *(const unsigned int*)(vh + (size_t)t * kDimHead + cw * 2);
    }
    *(unsigned int*)(sV + row * kDimHead + cw * 2) = val;
  }
#endif
  __syncthreads();

  const v8f zero8 = {0.f, 0.f, 0.f, 0.f, 0.f, 0.f, 0.f, 0.f};

  for (int mt = wave; mt < 13; mt += 4) {
    const int m0 = mt * 16;
    int qrow = m0 + mlane;
    if (qrow > kNseq - 1) qrow = kNseq - 1;  // clamp pad rows (never stored)
    const bf16_t* qrp = qh + (size_t)(1 + fidx * kNseq + qrow) * kDimHead;

    v16bf qa0, qa1;  // k-steps d=0..31 and d=32..63
#pragma unroll
    for (int v = 0; v < 8; ++v) {
      const int kk = hlf * 16 + v * 2;
      bf16x2 p0 = *(const bf16x2*)(qrp + kk);
      bf16x2 p1 = *(const bf16x2*)(qrp + 32 + kk);
      qa0[2 * v] = p0[0]; qa0[2 * v + 1] = p0[1];
      qa1[2 * v] = p1[0]; qa1[2 * v + 1] = p1[1];
    }

    // ---- S = q @ k^T over 13 key tiles -------------------------------
    v8f sacc[13];
#pragma unroll
    for (int nt = 0; nt < 13; ++nt) sacc[nt] = zero8;
#pragma unroll
    for (int nt = 0; nt < 13; ++nt) {
      const int key = nt * 16 + mlane;  // lane = N = key index
      v16bf kf0, kf1;
      if (key < kNkey) {
        const int t = (key == 0) ? 0 : (1 + fidx * kNseq + key - 1);
        const bf16_t* krp = kh + (size_t)t * kDimHead + hlf * 16;
#pragma unroll
        for (int v = 0; v < 8; ++v) {  // B K-dim = head dim: contiguous pairs
          bf16x2 p0 = *(const bf16x2*)(krp + 2 * v);
          bf16x2 p1 = *(const bf16x2*)(krp + 32 + 2 * v);
          kf0[2 * v] = p0[0]; kf0[2 * v + 1] = p0[1];
          kf1[2 * v] = p1[0]; kf1[2 * v + 1] = p1[1];
        }
      } else {
#pragma unroll
        for (int e = 0; e < 16; ++e) {
          kf0[e] = (bf16_t)0.f;
          kf1[e] = (bf16_t)0.f;
        }
      }
      sacc[nt] = wmma_bf16(qa0, kf0, sacc[nt]);
      sacc[nt] = wmma_bf16(qa1, kf1, sacc[nt]);
    }

    // ---- softmax rows (C layout: row = r + 8*hlf, col = nt*16 + mlane) ----
#pragma unroll
    for (int r = 0; r < 8; ++r) {
      float sv[13];
      float mx = -3.0e38f;
#pragma unroll
      for (int nt = 0; nt < 13; ++nt) {
        float xv = sacc[nt][r];
        if (nt * 16 + mlane >= kNkey) xv = -3.0e38f;  // mask pad columns
        sv[nt] = xv;
        mx = fmaxf(mx, xv);
      }
#pragma unroll
      for (int off = 1; off < 16; off <<= 1)
        mx = fmaxf(mx, __shfl_xor(mx, off, 32));
      float sum = 0.f;
#pragma unroll
      for (int nt = 0; nt < 13; ++nt) {
        float p = __expf(sv[nt] - mx);
        sv[nt] = p;
        sum += p;
      }
#pragma unroll
      for (int off = 1; off < 16; off <<= 1) sum += __shfl_xor(sum, off, 32);
      const float rinv = 1.f / sum;
      const int prow = r + 8 * hlf;
#pragma unroll
      for (int nt = 0; nt < 13; ++nt)
        sP[wave][prow * kNkeyP + nt * 16 + mlane] = (bf16_t)(sv[nt] * rinv);
      sP[wave][prow * kNkeyP + 208 + mlane] = (bf16_t)0.f;  // zero pad cols
    }

    // ---- O = P @ V (7 k-steps of 32 keys) ----------------------------
    v8f oacc[4];
#pragma unroll
    for (int j = 0; j < 4; ++j) oacc[j] = zero8;
#pragma unroll
    for (int ks = 0; ks < 7; ++ks) {
      v16bf pa;  // A fragment of P from per-wave LDS strip
      const bf16_t* prp = &sP[wave][mlane * kNkeyP + ks * 32];
#pragma unroll
      for (int v = 0; v < 8; ++v) {
        const int kk = hlf * 16 + v * 2;
        bf16x2 p = *(const bf16x2*)(prp + kk);
        pa[2 * v] = p[0];
        pa[2 * v + 1] = p[1];
      }
#pragma unroll
      for (int j = 0; j < 4; ++j) {
        v16bf vf;  // B fragment of V: lane = N = channel, K = key rows
        const bf16_t* vrp =
            sV + (size_t)(ks * 32 + hlf * 16) * kDimHead + j * 16 + mlane;
#pragma unroll
        for (int v = 0; v < 8; ++v) {
          vf[2 * v]     = vrp[(2 * v) * kDimHead];
          vf[2 * v + 1] = vrp[(2 * v + 1) * kDimHead];
        }
        oacc[j] = wmma_bf16(pa, vf, oacc[j]);
      }
    }

    // ---- store into [b][t][h*64 + d] bf16 ----------------------------
#pragma unroll
    for (int r = 0; r < 8; ++r) {
      const int row = m0 + r + 8 * hlf;
      if (row < kNseq) {
        const int t = 1 + fidx * kNseq + row;
        bf16_t* orow = attn + ((size_t)b * kNtok + t) * kDim + h * kDimHead;
#pragma unroll
        for (int j = 0; j < 4; ++j) orow[j * 16 + mlane] = (bf16_t)oacc[j][r];
      }
    }
  }
}

// ---------------------------------------------------------------------------
// Output GEMM: attn (12548 x 1024) bf16 @ woutT (1024 x 1024) bf16 -> fp32 out
// Same structure as QKV GEMM (async LDS B staging).
// ---------------------------------------------------------------------------
__global__ __launch_bounds__(128) void gemm_out_kernel(
    const bf16_t* __restrict__ attn, const bf16_t* __restrict__ wT,
    float* __restrict__ out) {
  __shared__ bf16_t sB[2][64 * kBStride];

  const int tid   = threadIdx.x;
  const int lane  = tid & 31;
  const int wave  = tid >> 5;
  const int mlane = lane & 15;
  const int hlf   = lane >> 4;
  const int row0  = blockIdx.x * 64 + wave * 16;
  const int col0  = blockIdx.y * 64;

  int arow = row0 + mlane;
  if (arow >= kM) arow = kM - 1;
  const bf16_t* aptr = attn + (size_t)arow * kDim;

  const v8f zero8 = {0.f, 0.f, 0.f, 0.f, 0.f, 0.f, 0.f, 0.f};
  v8f acc[4];
#pragma unroll
  for (int j = 0; j < 4; ++j) acc[j] = zero8;

  stage_b_tile(wT, col0, 0, sB[0], tid);

  for (int kt = 0; kt < kDim / 32; ++kt) {
    const int k0  = kt * 32;
    const int buf = kt & 1;
    if (kt + 1 < kDim / 32) {
      stage_b_tile(wT, col0, k0 + 32, sB[buf ^ 1], tid);
#if USE_ASYNC_LDS
      __builtin_amdgcn_s_wait_asynccnt(2);
#endif
    } else {
#if USE_ASYNC_LDS
      __builtin_amdgcn_s_wait_asynccnt(0);
#endif
    }
    __syncthreads();

    if (k0 + 256 < kDim) __builtin_prefetch(aptr + k0 + 256, 0, 0);
    v16bf afrag;
#pragma unroll
    for (int v = 0; v < 8; ++v) {
      const int kk = ((v & 4) ? 16 : 0) + hlf * 8 + (v & 3) * 2;
      bf16x2 p = *(const bf16x2*)(aptr + k0 + kk);
      afrag[2 * v] = p[0];
      afrag[2 * v + 1] = p[1];
    }
#pragma unroll
    for (int j = 0; j < 4; ++j) {
      v16bf bfrag = load_b_frag_lds(sB[buf], j * 16 + mlane, hlf);
      acc[j] = wmma_bf16(afrag, bfrag, acc[j]);
    }
    __syncthreads();
  }

#pragma unroll
  for (int j = 0; j < 4; ++j) {
    const int c = col0 + j * 16 + mlane;
#pragma unroll
    for (int r = 0; r < 8; ++r) {
      const int row = row0 + r + 8 * hlf;
      if (row < kM) out[(size_t)row * kDim + c] = acc[j][r];
    }
  }
}

// ---------------------------------------------------------------------------
extern "C" void kernel_launch(void* const* d_in, const int* in_sizes, int n_in,
                              void* d_out, int out_size, void* d_ws,
                              size_t ws_size, hipStream_t stream) {
  (void)in_sizes; (void)n_in; (void)out_size; (void)ws_size;
  const float* x     = (const float*)d_in[0];
  const float* w_qkv = (const float*)d_in[1];
  const float* w_out = (const float*)d_in[2];
  // d_in[3] is f (==16): baked in as kFrames.
  float* out = (float*)d_out;

  size_t off = 0;
  auto wsalloc = [&](size_t bytes) -> void* {
    void* p = (char*)d_ws + off;
    off += (bytes + 255) & ~(size_t)255;
    return p;
  };
  bf16_t* wqkvT = (bf16_t*)wsalloc((size_t)(3 * kDim) * kDim * sizeof(bf16_t));
  bf16_t* woutT = (bf16_t*)wsalloc((size_t)kDim * kDim * sizeof(bf16_t));
  const size_t hv = (size_t)kBatch * kHeads * kNtok * kDimHead;
  bf16_t* qbuf = (bf16_t*)wsalloc(hv * sizeof(bf16_t));
  bf16_t* kbuf = (bf16_t*)wsalloc(hv * sizeof(bf16_t));
  bf16_t* vbuf = (bf16_t*)wsalloc(hv * sizeof(bf16_t));
  bf16_t* abuf = (bf16_t*)wsalloc((size_t)kM * kDim * sizeof(bf16_t));

  // 1) weights -> transposed bf16
  transpose_w_kernel<<<dim3(kDim / 32, (3 * kDim) / 32), 256, 0, stream>>>(
      w_qkv, wqkvT, kDim, 3 * kDim);
  transpose_w_kernel<<<dim3(kDim / 32, kDim / 32), 256, 0, stream>>>(
      w_out, woutT, kDim, kDim);

  // 2) fused QKV projection (+ head split + q scale)
  gemm_qkv_kernel<<<dim3((kM + 63) / 64, (3 * kDim) / 64), 128, 0, stream>>>(
      x, wqkvT, qbuf, kbuf, vbuf);

  // 3) attention
  cls_attn_kernel<<<dim3(kBatch * kHeads), 64, 0, stream>>>(qbuf, kbuf, vbuf,
                                                            abuf);
  axial_attn_kernel<<<dim3(kBatch * kHeads * kFrames), 128, 0, stream>>>(
      qbuf, kbuf, vbuf, abuf);

  // 4) output projection -> fp32
  gemm_out_kernel<<<dim3((kM + 63) / 64, kDim / 64), 128, 0, stream>>>(
      abuf, woutT, out);
}
